// Model_61125974557080
// MI455X (gfx1250) — matrix operand order repaired and checked by reading, not compiled, not run on hardware
//
#include <hip/hip_runtime.h>

// ---------------------------------------------------------------------------
// MI455X (gfx1250) implementation. wave32; all GEMMs via v_wmma_f32_16x16x32_f16
// with pre-swizzled (fragment-layout) B operands, 16x64 tiles per wave, and a
// ping-pong (2x-unrolled, copy-free) software-pipelined inner loop.
// ---------------------------------------------------------------------------

typedef __attribute__((ext_vector_type(16))) _Float16 v16h;
typedef __attribute__((ext_vector_type(8)))  _Float16 v8h;
typedef __attribute__((ext_vector_type(8)))  float    v8f;
typedef _Float16 h16;

static constexpr int   kBatch = 4;
static constexpr int   kSeq   = 512;
static constexpr int   kCin   = 21;
static constexpr int   kTd    = 4;
static constexpr int   kDm    = 256;
static constexpr int   kH     = 8;
static constexpr int   kDh    = 32;
static constexpr int   kNK    = 4;
static constexpr int   kPL    = 16;
static constexpr int   kStr   = 8;
static constexpr int   kL     = 63;      // EFF_LEN
static constexpr int   kLpad  = 64;      // padded attn ld (TKA)
static constexpr int   kFF    = 1024;
static constexpr int   kPred  = 96;
static constexpr int   kNtok  = 260;     // 256 + 4 time dims
static constexpr int   kNtokPad = 272;   // padded attn ld (encoder), 16B-aligned rows
static constexpr float kEps   = 1e-5f;
static constexpr float kInvSqrtDh = 0.17677669529663687f;  // 1/sqrt(32)

#define DEV_INLINE __device__ __forceinline__

// host-side helpers for swizzled-B geometry
static inline int tnp(int N) { return ((((N + 15) / 16) + 3) / 4) * 4; }  // N tiles, padded to x4
static inline int kbl(int K) { return (K + 31) / 32; }                    // 32-wide K blocks

// ---------------------------------------------------------------------------
// A-fragment loaders (16-bit A 16x32 layout, CDNA5 ISA 7.12.2):
// lane holds row (lane&15); halves j=0..7 -> k = ka+j, j=8..15 -> k = ka+16+j.
// ---------------------------------------------------------------------------
DEV_INLINE v16h a_fast(const h16* __restrict__ ap, int ka)
{
    const v8h lo = *(const v8h*)(ap + ka);
    const v8h hv = *(const v8h*)(ap + ka + 16);
    v16h a;
#pragma unroll
    for (int j = 0; j < 8; ++j) { a[j] = lo[j]; a[j + 8] = hv[j]; }
    return a;
}

DEV_INLINE v16h a_tail(const h16* __restrict__ ap, int ka, int K)
{
    v16h a = {};
#pragma unroll
    for (int j = 0; j < 8; ++j) {
        if (ka + j      < K) a[j]     = ap[ka + j];
        if (ka + 16 + j < K) a[j + 8] = ap[ka + 16 + j];
    }
    return a;
}

// ---------------------------------------------------------------------------
// Generic batched WMMA GEMM:
//   C[M,N](f32) = A[M,K](f16 row-major) * B(K,N swizzled f16) + bias
// B is pre-swizzled into fragment chunks: chunk (z, tn, kb) = 32 lanes x 16
// contiguous halves; lane holds col tn*16+(lane&15), k = kb*32+(j>>3)*16+(lane>>4)*8+(j&7).
// One wave computes a 16x64 tile (A frag reused across 4 WMMAs per k-step).
// Out-of-range A rows are clamped (their outputs are never stored), keeping
// control flow wave-uniform. The inner loop is ping-pong double-buffered (2x
// unrolled) so next-block loads overlap WMMAs without register rotation
// copies. __launch_bounds__(128,1) lifts the VGPR budget so both fragment
// sets + 4 accumulators stay in registers (no scratch spills). The trailing
// preload may overread B by <=1 chunk: the workspace has a guard pad and the
// value is never consumed.
// ---------------------------------------------------------------------------
__global__ void __launch_bounds__(128, 1)
k_gemm(const h16* __restrict__ A, const h16* __restrict__ Bsw,
       float* __restrict__ C, const float* __restrict__ bias,
       int M, int N, int K, int lda, int ldc,
       int kblocks, int tilesNpad, int bBatched, int Hdim,
       long sAb, long sAh, long sCb, long sCh)
{
    const int z  = blockIdx.z;
    const int zb = z / Hdim, zh = z % Hdim;
    const h16* Ab = A + (long)zb * sAb + (long)zh * sAh;
    float*     Cb = C + (long)zb * sCb + (long)zh * sCh;

    const int tilesM  = (M + 15) >> 4;
    const int tilesN4 = tilesNpad >> 2;
    const int tile    = blockIdx.x * blockDim.y + threadIdx.y;   // wave-uniform
    if (tile >= tilesM * tilesN4) return;
    const int tm = tile / tilesN4;
    const int tg = tile % tilesN4;

    const int  lane  = threadIdx.x;
    const int  hi    = lane >> 4;
    const int  l16   = lane & 15;
    const int  arow  = tm * 16 + l16;
    // OOB rows clamped: they only corrupt output rows that are never stored.
    const int  arowC = (arow < M) ? arow : (M - 1);
    const h16* ap    = Ab + (long)arowC * lda;

    // swizzled-B: element for this lane of chunk c is bp[c*32]
    const v16h* bp = (const v16h*)Bsw + lane;
    const long  c0 = ((long)(bBatched ? z : 0) * tilesNpad + (long)tg * 4) * kblocks;

    const int kfull = K >> 5;                      // fully in-bounds 32-blocks
    const int nblk  = kfull + ((K & 31) ? 1 : 0);  // +1 ragged tail block

    v8f acc0 = {}, acc1 = {}, acc2 = {}, acc3 = {};
    v16h a0, b00, b01, b02, b03;   // ping buffer
    v16h a1, b10, b11, b12, b13;   // pong buffer

#define LOADF(Q, AV, B0, B1, B2, B3)                                           \
    {                                                                          \
        const int  q_  = (Q);                                                  \
        const int  ka_ = q_ * 32 + hi * 8;                                     \
        AV = (q_ < kfull) ? a_fast(ap, ka_) : a_tail(ap, ka_, K);              \
        const long cq_ = c0 + q_;                                              \
        B0 = bp[cq_ * 32];                                                     \
        B1 = bp[(cq_ + kblocks) * 32];                                         \
        B2 = bp[(cq_ + 2L * kblocks) * 32];                                    \
        B3 = bp[(cq_ + 3L * kblocks) * 32];                                    \
    }
#define WMMA4(AV, B0, B1, B2, B3)                                              \
    acc0 = __builtin_amdgcn_wmma_f32_16x16x32_f16(false, AV, false, B0, (short)0, acc0, false, false); \
    acc1 = __builtin_amdgcn_wmma_f32_16x16x32_f16(false, AV, false, B1, (short)0, acc1, false, false); \
    acc2 = __builtin_amdgcn_wmma_f32_16x16x32_f16(false, AV, false, B2, (short)0, acc2, false, false); \
    acc3 = __builtin_amdgcn_wmma_f32_16x16x32_f16(false, AV, false, B3, (short)0, acc3, false, false);

    LOADF(0, a0, b00, b01, b02, b03)
    // ping-pong main loop: two blocks per iteration, no buffer rotation copies
    for (int kb = 0; kb + 1 < nblk; kb += 2) {
        if ((kb + 4) * 32 < K) __builtin_prefetch(ap + (kb + 4) * 32, 0, 3);
        LOADF(kb + 1, a1, b10, b11, b12, b13)
        WMMA4(a0, b00, b01, b02, b03)
        LOADF(kb + 2, a0, b00, b01, b02, b03)   // overreads <=1 B chunk at end
        WMMA4(a1, b10, b11, b12, b13)
    }
    if (nblk & 1) { WMMA4(a0, b00, b01, b02, b03) }   // leftover odd block
#undef LOADF
#undef WMMA4

#define STORE_TILE(ACC, T)                                                     \
    {                                                                          \
        const int col = (tg * 4 + (T)) * 16 + l16;                             \
        if (col < N) {                                                         \
            const float bv = bias ? bias[col] : 0.f;                           \
            _Pragma("unroll")                                                  \
            for (int i = 0; i < 8; ++i) {                                      \
                const int row = tm * 16 + i + hi * 8;                          \
                if (row < M) Cb[(long)row * ldc + col] = ACC[i] + bv;          \
            }                                                                  \
        }                                                                      \
    }
    STORE_TILE(acc0, 0)
    STORE_TILE(acc1, 1)
    STORE_TILE(acc2, 2)
    STORE_TILE(acc3, 3)
#undef STORE_TILE
}

// ----------------------- swizzled-B staging kernels -------------------------

DEV_INLINE void swiz_decode(long i, int kblocks, int tilesNpad,
                            int& z, int& k, int& n)
{
    const int j    = (int)(i & 15);
    const int lane = (int)((i >> 4) & 31);
    const long chunk = i >> 9;
    const int kb = (int)(chunk % kblocks);
    const long tnAll = chunk / kblocks;
    const int tn = (int)(tnAll % tilesNpad);
    z = (int)(tnAll / tilesNpad);
    k = kb * 32 + (j & 8) * 2 + (lane >> 4) * 8 + (j & 7);
    n = tn * 16 + (lane & 15);
}

// Weight W[N,K] f32 -> swizzled B(K,N); single batch.
__global__ void k_swiz_w(const float* __restrict__ W, h16* __restrict__ out,
                         int N, int K, int kblocks, int tilesNpad, long total)
{
    const long i = (long)blockIdx.x * blockDim.x + threadIdx.x;
    if (i >= total) return;
    int z, k, n; swiz_decode(i, kblocks, tilesNpad, z, k, n);
    float v = 0.f;
    if (k < K && n < N) v = W[(long)n * K + k];
    out[i] = (h16)v;
}

// patch_W (256,256,16) f32 -> swizzled B with k = kp*256 + din, n = dout.
__global__ void k_swiz_patch(const float* __restrict__ W, h16* __restrict__ out)
{
    const long total = (long)16 * 128 * 512;  // tilesNpad=16, kblocks=128
    const long i = (long)blockIdx.x * blockDim.x + threadIdx.x;
    if (i >= total) return;
    int z, k, n; swiz_decode(i, 128, 16, z, k, n);
    const int kp = k >> 8, din = k & 255;
    out[i] = (h16)W[((long)n * kDm + din) * kPL + kp];
}

// Per-(b,h) K^T / Xh^T: B[k=d][n=j] = X[b, j, h*32+d].  K=32, N=L.  32 batches.
__global__ void k_swiz_headT(const float* __restrict__ X, h16* __restrict__ out,
                             int L, int tilesNpad, long total)
{
    const long i = (long)blockIdx.x * blockDim.x + threadIdx.x;
    if (i >= total) return;
    int z, k, n; swiz_decode(i, 1, tilesNpad, z, k, n);
    const int b = z / kH, h = z % kH;
    float v = 0.f;
    if (k < kDh && n < L) v = X[((long)b * L + n) * kDm + h * kDh + k];
    out[i] = (h16)v;
}

// Per-(b,h) V: B[k=j][n=d] = X[b, j, h*32+d].  K=L, N=32.  32 batches.
__global__ void k_swiz_headV(const float* __restrict__ X, h16* __restrict__ out,
                             int L, int kblocks, int tilesNpad, long total)
{
    const long i = (long)blockIdx.x * blockDim.x + threadIdx.x;
    if (i >= total) return;
    int z, k, n; swiz_decode(i, kblocks, tilesNpad, z, k, n);
    const int b = z / kH, h = z % kH;
    float v = 0.f;
    if (k < L && n < kDh) v = X[((long)b * L + k) * kDm + h * kDh + n];
    out[i] = (h16)v;
}

// ------------------------- Pre/post-processing kernels ----------------------

__global__ void k_stats(const float* __restrict__ x, float* __restrict__ mean,
                        float* __restrict__ stdev)
{
    const int bc = blockIdx.x;
    const int b  = bc / kCin, c = bc % kCin;
    __shared__ float s1[256], s2[256];
    float a = 0.f, q = 0.f;
    for (int s = threadIdx.x; s < kSeq; s += 256) {
        const float v = x[((long)b * kSeq + s) * kCin + c];
        a += v; q += v * v;
    }
    s1[threadIdx.x] = a; s2[threadIdx.x] = q;
    __syncthreads();
    for (int o = 128; o > 0; o >>= 1) {
        if (threadIdx.x < o) { s1[threadIdx.x] += s1[threadIdx.x + o];
                               s2[threadIdx.x] += s2[threadIdx.x + o]; }
        __syncthreads();
    }
    if (threadIdx.x == 0) {
        const float m = s1[0] / (float)kSeq;
        const float v = s2[0] / (float)kSeq - m * m;
        mean[bc]  = m;
        stdev[bc] = sqrtf(v + kEps);
    }
}

__global__ void k_token_embed(const float* __restrict__ x, const float* __restrict__ mean,
                              const float* __restrict__ stdev, const float* __restrict__ tokW,
                              h16* __restrict__ enc0)
{
    const int s = blockIdx.x, b = blockIdx.y, d = threadIdx.x;
    __shared__ float xs[3 * kCin];
    if (d < 3 * kCin) {
        const int kk = d / kCin, c = d % kCin;
        const int sidx = (s - 1 + kk + kSeq) % kSeq;
        xs[d] = (x[((long)b * kSeq + sidx) * kCin + c] - mean[b * kCin + c]) /
                stdev[b * kCin + c];
    }
    __syncthreads();
    float acc = 0.f;
#pragma unroll
    for (int kk = 0; kk < 3; ++kk)
        for (int c = 0; c < kCin; ++c)
            acc += xs[kk * kCin + c] * tokW[(d * kCin + c) * 3 + kk];
    const int j = d >> 1;
    const float div = __expf((float)(2 * j) * (-9.210340371976184f / 256.0f));
    const float ang = (float)s * div;
    acc += (d & 1) ? __cosf(ang) : __sinf(ang);
    enc0[((long)b * kSeq + s) * kDm + d] = (h16)acc;
}

__global__ void k_time_avg(const float* __restrict__ xm, float* __restrict__ t)
{
    const int idx = blockIdx.x * blockDim.x + threadIdx.x;
    if (idx >= kBatch * kL * kTd) return;
    const int tt = idx % kTd;
    const int i  = (idx / kTd) % kL;
    const int b  = idx / (kTd * kL);
    float s = 0.f;
    for (int p = 0; p < kPL; ++p)
        s += xm[((long)b * kSeq + i * kStr + p) * kTd + tt];
    t[idx] = s * (1.0f / kPL);
}

__global__ void k_cvt_f16(const float* __restrict__ src, h16* __restrict__ dst, long n)
{
    const long i = (long)blockIdx.x * blockDim.x + threadIdx.x;
    if (i < n) dst[i] = (h16)src[i];
}

// gi[b,h,i] = mean_dh sum_{t,k} w*exp(-(t-mu)^2/(2 sg^2))
__global__ void k_gi(const float* __restrict__ t, const float* __restrict__ mu,
                     const float* __restrict__ sg, const float* __restrict__ w,
                     float* __restrict__ gi, int layer)
{
    const int idx = blockIdx.x * blockDim.x + threadIdx.x;
    if (idx >= kBatch * kH * kL) return;
    const int i = idx % kL;
    const int h = (idx / kL) % kH;
    const int b = idx / (kL * kH);
    const float* tv = t + ((long)b * kL + i) * kTd;
    const int base = (layer * kH + h) * kDh * kTd * kNK;
    float s = 0.f;
    for (int dh = 0; dh < kDh; ++dh)
        for (int tt = 0; tt < kTd; ++tt) {
            const float tx = tv[tt];
            for (int kk = 0; kk < kNK; ++kk) {
                const int p = base + (dh * kTd + tt) * kNK + kk;
                const float diff = tx - mu[p];
                const float s2   = sg[p];
                s += w[p] * __expf(-diff * diff / (2.f * s2 * s2));
            }
        }
    gi[idx] = s * (1.0f / kDh);
}

// sc = qk / sqrt(Dh) * (alpha*P*2*gi + beta*P*g_rel + gamma)   (in-place ok)
__global__ void k_tka_score(const float* __restrict__ qk, const float* __restrict__ Pij,
                            const float* __restrict__ gi, const float* __restrict__ t,
                            const float* __restrict__ mu_r, const float* __restrict__ sg_r,
                            const float* __restrict__ w_r, const float* __restrict__ alpha,
                            const float* __restrict__ beta, const float* __restrict__ gamma,
                            float* __restrict__ sc, int layer)
{
    const int bh = blockIdx.z;
    const int b = bh / kH, h = bh % kH;
    const int ij = blockIdx.x * blockDim.x + threadIdx.x;
    if (ij >= kL * kL) return;
    const int i = ij / kL, j = ij % kL;
    float tr[kTd];
#pragma unroll
    for (int tt = 0; tt < kTd; ++tt)
        tr[tt] = fabsf(t[((long)b * kL + i) * kTd + tt] - t[((long)b * kL + j) * kTd + tt]);
    const int base = (layer * kH + h) * kDh * kTd * kNK;
    float grel = 0.f;
    for (int dh = 0; dh < kDh; ++dh)
        for (int tt = 0; tt < kTd; ++tt) {
            const float tx = tr[tt];
            for (int kk = 0; kk < kNK; ++kk) {
                const int p = base + (dh * kTd + tt) * kNK + kk;
                const float diff = tx - mu_r[p];
                const float s2   = sg_r[p];
                grel += w_r[p] * __expf(-diff * diff / (2.f * s2 * s2));
            }
        }
    grel *= (1.0f / kDh);
    const long off = (long)bh * kL * kL + ij;
    const float P  = Pij[off];
    const int   lh = layer * kH + h;
    const float A  = alpha[lh] * P * 2.0f * gi[(b * kH + h) * kL + i] +
                     beta[lh]  * P * grel + gamma[lh];
    sc[off] = qk[off] * kInvSqrtDh * A;
}

// Row softmax, f16 output with padded leading dimension. One wave per row.
__global__ void k_softmax(const float* __restrict__ S, h16* __restrict__ P,
                          int cols, int ldp, float scale)
{
    const int r = blockIdx.x;
    const int lane = threadIdx.x;
    const float* s = S + (long)r * cols;
    h16* p = P + (long)r * ldp;
    float mx = -1e30f;
    for (int c = lane; c < cols; c += 32) mx = fmaxf(mx, s[c] * scale);
    for (int o = 16; o > 0; o >>= 1) mx = fmaxf(mx, __shfl_xor(mx, o, 32));
    float sum = 0.f;
    for (int c = lane; c < cols; c += 32) sum += __expf(s[c] * scale - mx);
    for (int o = 16; o > 0; o >>= 1) sum += __shfl_xor(sum, o, 32);
    const float inv = 1.0f / sum;
    for (int c = lane; c < cols; c += 32)
        p[c] = (h16)(__expf(s[c] * scale - mx) * inv);
}

// out = LayerNorm(X (+ Y)) * g + b ; optional f32 and f16 outputs.
__global__ void k_add_ln(const float* __restrict__ X, const float* __restrict__ Y,
                         const float* __restrict__ g, const float* __restrict__ bt,
                         float* __restrict__ out32, h16* __restrict__ out16)
{
    const int r = blockIdx.x, d = threadIdx.x;   // 256 threads
    __shared__ float red[256];
    __shared__ float mv[2];
    float v = X[(long)r * kDm + d];
    if (Y) v += Y[(long)r * kDm + d];
    red[d] = v; __syncthreads();
    for (int o = 128; o > 0; o >>= 1) { if (d < o) red[d] += red[d + o]; __syncthreads(); }
    if (d == 0) mv[0] = red[0] * (1.0f / kDm);
    __syncthreads();
    const float m = mv[0];
    const float c = v - m;
    red[d] = c * c; __syncthreads();
    for (int o = 128; o > 0; o >>= 1) { if (d < o) red[d] += red[d + o]; __syncthreads(); }
    if (d == 0) mv[1] = red[0] * (1.0f / kDm);
    __syncthreads();
    const float o_ = c * rsqrtf(mv[1] + kEps) * g[d] + bt[d];
    if (out32) out32[(long)r * kDm + d] = o_;
    if (out16) out16[(long)r * kDm + d] = (h16)o_;
}

__global__ void k_gelu(const float* __restrict__ x, h16* __restrict__ y, long n)
{
    const long i = (long)blockIdx.x * blockDim.x + threadIdx.x;
    if (i >= n) return;
    const float v = x[i];
    y[i] = (h16)(0.5f * v * (1.0f + erff(v * 0.7071067811865476f)));
}

// xi[b, n, k] = n<256 ? enc[b,k,n] : t[b,k,n-256]  (K padded to 64)
__global__ void k_build_xi(const float* __restrict__ enc, const float* __restrict__ t,
                           h16* __restrict__ xi)
{
    const long i = (long)blockIdx.x * blockDim.x + threadIdx.x;
    if (i >= (long)kBatch * kNtok * 64) return;
    const int k = (int)(i % 64);
    const int n = (int)((i / 64) % kNtok);
    const int b = (int)(i / (64L * kNtok));
    float v = 0.f;
    if (k < kL) v = (n < kDm) ? enc[((long)b * kL + k) * kDm + n]
                              : t[((long)b * kL + k) * kTd + (n - kDm)];
    xi[i] = (h16)v;
}

// out[b,p,c] = P[b, token=c, p] * stdev[b,c] + mean[b,c]
__global__ void k_out(const float* __restrict__ P, const float* __restrict__ mean,
                      const float* __restrict__ stdev, float* __restrict__ out)
{
    const int i = blockIdx.x * blockDim.x + threadIdx.x;
    if (i >= kBatch * kPred * kCin) return;
    const int c = i % kCin;
    const int p = (i / kCin) % kPred;
    const int b = i / (kCin * kPred);
    out[i] = P[((long)b * kNtok + c) * kPred + p] * stdev[b * kCin + c] + mean[b * kCin + c];
}

// ----------------------------- host-side helpers ----------------------------

static inline void launch_gemm(hipStream_t st, const h16* A, const h16* Bsw, float* C,
                               const float* bias, int M, int N, int K,
                               int lda, int ldc, int kblocks, int tilesNpad,
                               int bBatched, int batch, int Hdim,
                               long sAb, long sAh, long sCb, long sCh)
{
    const int tilesM  = (M + 15) / 16;
    const int tilesN4 = tilesNpad / 4;
    const int tiles   = tilesM * tilesN4;
    dim3 blk(32, 4, 1), grd((unsigned)((tiles + 3) / 4), 1, (unsigned)batch);
    k_gemm<<<grd, blk, 0, st>>>(A, Bsw, C, bias, M, N, K, lda, ldc,
                                kblocks, tilesNpad, bBatched, Hdim,
                                sAb, sAh, sCb, sCh);
}

static inline unsigned ew(long n, int bs = 256) { return (unsigned)((n + bs - 1) / bs); }
static inline long swizHalves(int Z, int N, int K) { return (long)Z * tnp(N) * kbl(K) * 512; }

extern "C" void kernel_launch(void* const* d_in, const int* in_sizes, int n_in,
                              void* d_out, int out_size, void* d_ws, size_t ws_size,
                              hipStream_t stream)
{
    // ---- inputs (setup_inputs order) ----
    const float* x_enc   = (const float*)d_in[0];
    const float* x_mark  = (const float*)d_in[1];
    const float* tok_W   = (const float*)d_in[4];
    const float* patch_W = (const float*)d_in[5];
    const float* patch_b = (const float*)d_in[6];
    const float* nl_WQ   = (const float*)d_in[7];
    const float* nl_WK   = (const float*)d_in[8];
    const float* nl_WV   = (const float*)d_in[9];
    const float* nl_WO   = (const float*)d_in[10];
    const float* nl_bO   = (const float*)d_in[11];
    const float* nl_mu_a = (const float*)d_in[12];
    const float* nl_sg_a = (const float*)d_in[13];
    const float* nl_w_a  = (const float*)d_in[14];
    const float* nl_mu_r = (const float*)d_in[15];
    const float* nl_sg_r = (const float*)d_in[16];
    const float* nl_w_r  = (const float*)d_in[17];
    const float* nl_al   = (const float*)d_in[18];
    const float* nl_be   = (const float*)d_in[19];
    const float* nl_ga   = (const float*)d_in[20];
    const float* inv_W   = (const float*)d_in[21];
    const float* inv_b   = (const float*)d_in[22];
    const float* el_Wq   = (const float*)d_in[23];
    const float* el_bq   = (const float*)d_in[24];
    const float* el_Wk   = (const float*)d_in[25];
    const float* el_bk   = (const float*)d_in[26];
    const float* el_Wv   = (const float*)d_in[27];
    const float* el_bv   = (const float*)d_in[28];
    const float* el_Wo   = (const float*)d_in[29];
    const float* el_bo   = (const float*)d_in[30];
    const float* el_W1   = (const float*)d_in[31];
    const float* el_b1   = (const float*)d_in[32];
    const float* el_W2   = (const float*)d_in[33];
    const float* el_b2   = (const float*)d_in[34];
    const float* n1_g    = (const float*)d_in[35];
    const float* n1_b    = (const float*)d_in[36];
    const float* n2_g    = (const float*)d_in[37];
    const float* n2_b    = (const float*)d_in[38];
    const float* norm_g  = (const float*)d_in[39];
    const float* norm_b  = (const float*)d_in[40];
    const float* proj_W  = (const float*)d_in[41];
    const float* proj_b  = (const float*)d_in[42];
    float* out = (float*)d_out;

    // ---- workspace layout (bump allocator) ----
    char* ws = (char*)d_ws;
    size_t off = 0;
    auto alloc = [&](size_t bytes) -> void* {
        off = (off + 255) & ~(size_t)255;
        void* p = ws + off; off += bytes; return p;
    };
    const int MA = kBatch * kL;      // 252 TKA tokens
    const int MC = kBatch * kNtok;   // 1040 variate tokens
    const int BH = kBatch * kH;      // 32 (b,h) pairs

    float* mean   = (float*)alloc(kBatch * kCin * 4);
    float* stdev  = (float*)alloc(kBatch * kCin * 4);
    float* tf     = (float*)alloc((size_t)kBatch * kL * kTd * 4);
    float* encA32 = (float*)alloc((size_t)MA * kDm * 4);
    float* enc232 = (float*)alloc((size_t)MC * kDm * 4);
    float* q32    = (float*)alloc((size_t)MC * kDm * 4);
    float* k32    = (float*)alloc((size_t)MC * kDm * 4);
    float* v32    = (float*)alloc((size_t)MC * kDm * 4);
    float* Pij32  = (float*)alloc((size_t)BH * kL * kL * 4);
    float* sc32   = (float*)alloc((size_t)BH * kNtok * kNtok * 4);
    float* gi     = (float*)alloc((size_t)BH * kL * 4);
    float* ctx32  = (float*)alloc((size_t)MC * kDm * 4);
    float* o32    = (float*)alloc((size_t)MC * kDm * 4);
    float* xn32   = (float*)alloc((size_t)MC * kDm * 4);
    float* ffn32  = (float*)alloc((size_t)MC * kFF * 4);
    float* y232   = (float*)alloc((size_t)MC * kDm * 4);
    float* P32    = (float*)alloc((size_t)MC * kPred * 4);

    h16* enc0h  = (h16*)alloc((size_t)kBatch * kSeq * kDm * 2);
    h16* pWt    = (h16*)alloc((size_t)swizHalves(1, kDm, kPL * kDm) * 2);
    h16* Wt0    = (h16*)alloc((size_t)swizHalves(1, kDm, kDm) * 2);
    h16* Wt1    = (h16*)alloc((size_t)swizHalves(1, kDm, kDm) * 2);
    h16* Wt2    = (h16*)alloc((size_t)swizHalves(1, kDm, kDm) * 2);
    h16* Wt3    = (h16*)alloc((size_t)swizHalves(1, kDm, kDm) * 2);
    h16* W1t    = (h16*)alloc((size_t)swizHalves(1, kFF, kDm) * 2);
    h16* W2t    = (h16*)alloc((size_t)swizHalves(1, kDm, kFF) * 2);
    h16* invWt  = (h16*)alloc((size_t)swizHalves(1, kDm, kL) * 2);
    h16* projt  = (h16*)alloc((size_t)swizHalves(1, kPred, kDm) * 2);
    h16* encTsw = (h16*)alloc((size_t)swizHalves(BH, kL, kDh) * 2);
    h16* kTsw   = (h16*)alloc((size_t)swizHalves(BH, kNtok, kDh) * 2);   // max of the two uses
    h16* vSw    = (h16*)alloc((size_t)swizHalves(BH, kDh, kNtok) * 2);   // max of the two uses
    h16* encAh  = (h16*)alloc((size_t)MC * kDm * 2);
    h16* qh     = (h16*)alloc((size_t)MC * kDm * 2);
    h16* attnh  = (h16*)alloc((size_t)BH * kNtok * kNtokPad * 2);        // padded lds
    h16* ctxh   = (h16*)alloc((size_t)MC * kDm * 2);
    h16* xnh    = (h16*)alloc((size_t)MC * kDm * 2);
    h16* ffnh   = (h16*)alloc((size_t)MC * kFF * 2);
    h16* xih    = (h16*)alloc((size_t)MC * 64 * 2);
    (void)alloc(8192);   // guard pad: ping-pong preload may overread <=1 chunk
    (void)ws_size; (void)in_sizes; (void)n_in; (void)out_size;

    // ---- stage 1: instance norm stats, token embed + PE, patch times ----
    k_stats<<<kBatch * kCin, 256, 0, stream>>>(x_enc, mean, stdev);
    k_token_embed<<<dim3(kSeq, kBatch), kDm, 0, stream>>>(x_enc, mean, stdev, tok_W, enc0h);
    k_time_avg<<<ew(kBatch * kL * kTd), 256, 0, stream>>>(x_mark, tf);

    // ---- stage 2: patch conv as im2col GEMM (M=63/batch, K=4096, N=256) ----
    k_swiz_patch<<<ew(swizHalves(1, kDm, kPL * kDm)), 256, 0, stream>>>(patch_W, pWt);
    launch_gemm(stream, enc0h, pWt, encA32, patch_b,
                kL, kDm, kPL * kDm, kStr * kDm /*lda: stride-8 windows*/, kDm,
                kbl(kPL * kDm), tnp(kDm), 0, kBatch, 1,
                (long)kSeq * kDm, 0, (long)kL * kDm, 0);

    // ---- stage 3: two TKA layers ----
    for (int l = 0; l < 2; ++l) {
        k_cvt_f16<<<ew((long)MA * kDm), 256, 0, stream>>>(encA32, encAh, (long)MA * kDm);
        k_swiz_headT<<<ew(swizHalves(BH, kL, kDh)), 256, 0, stream>>>(
            encA32, encTsw, kL, tnp(kL), swizHalves(BH, kL, kDh));
        k_swiz_w<<<ew(swizHalves(1, kDm, kDm)), 256, 0, stream>>>(
            nl_WQ + (long)l * kDm * kDm, Wt0, kDm, kDm, kbl(kDm), tnp(kDm), swizHalves(1, kDm, kDm));
        k_swiz_w<<<ew(swizHalves(1, kDm, kDm)), 256, 0, stream>>>(
            nl_WK + (long)l * kDm * kDm, Wt1, kDm, kDm, kbl(kDm), tnp(kDm), swizHalves(1, kDm, kDm));
        k_swiz_w<<<ew(swizHalves(1, kDm, kDm)), 256, 0, stream>>>(
            nl_WV + (long)l * kDm * kDm, Wt2, kDm, kDm, kbl(kDm), tnp(kDm), swizHalves(1, kDm, kDm));
        k_swiz_w<<<ew(swizHalves(1, kDm, kDm)), 256, 0, stream>>>(
            nl_WO + (long)l * kDm * kDm, Wt3, kDm, kDm, kbl(kDm), tnp(kDm), swizHalves(1, kDm, kDm));
        // Pij = Xh @ Xh^T per (b,h)
        launch_gemm(stream, encAh, encTsw, Pij32, nullptr, kL, kL, kDh, kDm, kL,
                    kbl(kDh), tnp(kL), 1, BH, kH,
                    (long)kL * kDm, kDh, (long)kH * kL * kL, (long)kL * kL);
        // Q,K,V projections (no bias)
        launch_gemm(stream, encAh, Wt0, q32, nullptr, MA, kDm, kDm, kDm, kDm,
                    kbl(kDm), tnp(kDm), 0, 1, 1, 0, 0, 0, 0);
        launch_gemm(stream, encAh, Wt1, k32, nullptr, MA, kDm, kDm, kDm, kDm,
                    kbl(kDm), tnp(kDm), 0, 1, 1, 0, 0, 0, 0);
        launch_gemm(stream, encAh, Wt2, v32, nullptr, MA, kDm, kDm, kDm, kDm,
                    kbl(kDm), tnp(kDm), 0, 1, 1, 0, 0, 0, 0);
        k_cvt_f16<<<ew((long)MA * kDm), 256, 0, stream>>>(q32, qh, (long)MA * kDm);
        k_swiz_headT<<<ew(swizHalves(BH, kL, kDh)), 256, 0, stream>>>(
            k32, kTsw, kL, tnp(kL), swizHalves(BH, kL, kDh));
        k_swiz_headV<<<ew(swizHalves(BH, kDh, kL)), 256, 0, stream>>>(
            v32, vSw, kL, kbl(kL), tnp(kDh), swizHalves(BH, kDh, kL));
        // qk = Q @ K^T per (b,h)
        launch_gemm(stream, qh, kTsw, sc32, nullptr, kL, kL, kDh, kDm, kL,
                    kbl(kDh), tnp(kL), 1, BH, kH,
                    (long)kL * kDm, kDh, (long)kH * kL * kL, (long)kL * kL);
        // Gaussian modulation + scale, then softmax
        k_gi<<<ew(kBatch * kH * kL), 256, 0, stream>>>(tf, nl_mu_a, nl_sg_a, nl_w_a, gi, l);
        k_tka_score<<<dim3(ew(kL * kL), 1, BH), 256, 0, stream>>>(
            sc32, Pij32, gi, tf, nl_mu_r, nl_sg_r, nl_w_r, nl_al, nl_be, nl_ga, sc32, l);
        k_softmax<<<BH * kL, 32, 0, stream>>>(sc32, attnh, kL, kLpad, 1.0f);
        // ctx = attn @ V per (b,h), merged into (252,256)
        launch_gemm(stream, attnh, vSw, ctx32, nullptr, kL, kDh, kL, kLpad, kDm,
                    kbl(kL), tnp(kDh), 1, BH, kH,
                    (long)kH * kL * kLpad, (long)kL * kLpad, (long)kL * kDm, kDh);
        k_cvt_f16<<<ew((long)MA * kDm), 256, 0, stream>>>(ctx32, ctxh, (long)MA * kDm);
        // output projection (+bO) -> next enc
        launch_gemm(stream, ctxh, Wt3, encA32, nl_bO + (long)l * kDm,
                    MA, kDm, kDm, kDm, kDm, kbl(kDm), tnp(kDm), 0, 1, 1, 0, 0, 0, 0);
    }

    // ---- stage 4: inverted embedding (B,260,63) @ inv_W^T ----
    k_build_xi<<<ew((long)MC * 64), 256, 0, stream>>>(encA32, tf, xih);
    k_swiz_w<<<ew(swizHalves(1, kDm, kL)), 256, 0, stream>>>(
        inv_W, invWt, kDm, kL, kbl(kL), tnp(kDm), swizHalves(1, kDm, kL));
    launch_gemm(stream, xih, invWt, enc232, inv_b, MC, kDm, kL, 64, kDm,
                kbl(kL), tnp(kDm), 0, 1, 1, 0, 0, 0, 0);

    // ---- stage 5: two standard encoder layers ----
    for (int i = 0; i < 2; ++i) {
        k_cvt_f16<<<ew((long)MC * kDm), 256, 0, stream>>>(enc232, encAh, (long)MC * kDm);
        k_swiz_w<<<ew(swizHalves(1, kDm, kDm)), 256, 0, stream>>>(
            el_Wq + (long)i * kDm * kDm, Wt0, kDm, kDm, kbl(kDm), tnp(kDm), swizHalves(1, kDm, kDm));
        k_swiz_w<<<ew(swizHalves(1, kDm, kDm)), 256, 0, stream>>>(
            el_Wk + (long)i * kDm * kDm, Wt1, kDm, kDm, kbl(kDm), tnp(kDm), swizHalves(1, kDm, kDm));
        k_swiz_w<<<ew(swizHalves(1, kDm, kDm)), 256, 0, stream>>>(
            el_Wv + (long)i * kDm * kDm, Wt2, kDm, kDm, kbl(kDm), tnp(kDm), swizHalves(1, kDm, kDm));
        k_swiz_w<<<ew(swizHalves(1, kDm, kDm)), 256, 0, stream>>>(
            el_Wo + (long)i * kDm * kDm, Wt3, kDm, kDm, kbl(kDm), tnp(kDm), swizHalves(1, kDm, kDm));
        k_swiz_w<<<ew(swizHalves(1, kFF, kDm)), 256, 0, stream>>>(
            el_W1 + (long)i * kFF * kDm, W1t, kFF, kDm, kbl(kDm), tnp(kFF), swizHalves(1, kFF, kDm));
        k_swiz_w<<<ew(swizHalves(1, kDm, kFF)), 256, 0, stream>>>(
            el_W2 + (long)i * kDm * kFF, W2t, kDm, kFF, kbl(kFF), tnp(kDm), swizHalves(1, kDm, kFF));
        launch_gemm(stream, encAh, Wt0, q32, el_bq + (long)i * kDm,
                    MC, kDm, kDm, kDm, kDm, kbl(kDm), tnp(kDm), 0, 1, 1, 0, 0, 0, 0);
        launch_gemm(stream, encAh, Wt1, k32, el_bk + (long)i * kDm,
                    MC, kDm, kDm, kDm, kDm, kbl(kDm), tnp(kDm), 0, 1, 1, 0, 0, 0, 0);
        launch_gemm(stream, encAh, Wt2, v32, el_bv + (long)i * kDm,
                    MC, kDm, kDm, kDm, kDm, kbl(kDm), tnp(kDm), 0, 1, 1, 0, 0, 0, 0);
        k_cvt_f16<<<ew((long)MC * kDm), 256, 0, stream>>>(q32, qh, (long)MC * kDm);
        k_swiz_headT<<<ew(swizHalves(BH, kNtok, kDh)), 256, 0, stream>>>(
            k32, kTsw, kNtok, tnp(kNtok), swizHalves(BH, kNtok, kDh));
        k_swiz_headV<<<ew(swizHalves(BH, kDh, kNtok)), 256, 0, stream>>>(
            v32, vSw, kNtok, kbl(kNtok), tnp(kDh), swizHalves(BH, kDh, kNtok));
        // scores (b,h): (260,32)x(32,260)
        launch_gemm(stream, qh, kTsw, sc32, nullptr, kNtok, kNtok, kDh, kDm, kNtok,
                    kbl(kDh), tnp(kNtok), 1, BH, kH,
                    (long)kNtok * kDm, kDh, (long)kH * kNtok * kNtok, (long)kNtok * kNtok);
        k_softmax<<<BH * kNtok, 32, 0, stream>>>(sc32, attnh, kNtok, kNtokPad, kInvSqrtDh);
        // ctx (b,h): (260,260)x(260,32) merged into (1040,256)
        launch_gemm(stream, attnh, vSw, ctx32, nullptr, kNtok, kDh, kNtok, kNtokPad, kDm,
                    kbl(kNtok), tnp(kDh), 1, BH, kH,
                    (long)kH * kNtok * kNtokPad, (long)kNtok * kNtokPad,
                    (long)kNtok * kDm, kDh);
        k_cvt_f16<<<ew((long)MC * kDm), 256, 0, stream>>>(ctx32, ctxh, (long)MC * kDm);
        launch_gemm(stream, ctxh, Wt3, o32, el_bo + (long)i * kDm,
                    MC, kDm, kDm, kDm, kDm, kbl(kDm), tnp(kDm), 0, 1, 1, 0, 0, 0, 0);
        k_add_ln<<<MC, kDm, 0, stream>>>(enc232, o32, n1_g + (long)i * kDm,
                                         n1_b + (long)i * kDm, xn32, xnh);
        launch_gemm(stream, xnh, W1t, ffn32, el_b1 + (long)i * kFF,
                    MC, kFF, kDm, kDm, kFF, kbl(kDm), tnp(kFF), 0, 1, 1, 0, 0, 0, 0);
        k_gelu<<<ew((long)MC * kFF), 256, 0, stream>>>(ffn32, ffnh, (long)MC * kFF);
        launch_gemm(stream, ffnh, W2t, y232, el_b2 + (long)i * kDm,
                    MC, kDm, kFF, kFF, kDm, kbl(kFF), tnp(kDm), 0, 1, 1, 0, 0, 0, 0);
        k_add_ln<<<MC, kDm, 0, stream>>>(xn32, y232, n2_g + (long)i * kDm,
                                         n2_b + (long)i * kDm, enc232, encAh);
    }

    // ---- stage 6: final LN, projection, de-norm gather ----
    k_add_ln<<<MC, kDm, 0, stream>>>(enc232, nullptr, norm_g, norm_b, nullptr, encAh);
    k_swiz_w<<<ew(swizHalves(1, kPred, kDm)), 256, 0, stream>>>(
        proj_W, projt, kPred, kDm, kbl(kDm), tnp(kPred), swizHalves(1, kPred, kDm));
    launch_gemm(stream, encAh, projt, P32, proj_b, MC, kPred, kDm, kDm, kPred,
                kbl(kDm), tnp(kPred), 0, 1, 1, 0, 0, 0, 0);
    k_out<<<ew(kBatch * kPred * kCin), 256, 0, stream>>>(P32, mean, stdev, out);
}
